// ResidualMemory_28157805593470
// MI455X (gfx1250) — compile-verified, matrix operand order checked by reading, not run on hardware
//
#include <hip/hip_runtime.h>

// ---------------------------------------------------------------------------
// CDNA5 (gfx1250, wave32) fused ResidualMemory kernel: bf16 WMMA + TDM
// double-buffered weight streaming (tensor_load_to_lds / TENSORcnt).
// ---------------------------------------------------------------------------

typedef __attribute__((ext_vector_type(16))) __bf16       v16bf;
typedef __attribute__((ext_vector_type(8)))  float        v8f;
typedef __attribute__((ext_vector_type(4)))  unsigned int u32x4;
typedef __attribute__((ext_vector_type(8)))  unsigned int u32x8;

#define C_DIM    256
#define L_DIM    4096
#define POS_TILE 64        // positions per workgroup (4 m-tiles)
#define SQF      16
#define NMEM     64
#define XSTR     264       // 256 + 8 pad (bf16 elements) == TDM padded row stride
#define PSTR     72        // 64 + 8 pad
#define CH_ROWS  64        // weight rows per TDM chunk
#define N_CHUNK  16        // 1024 / 64

// d_ws layout, in ushort (bf16) elements
#define WQ_OFF   0         // 1024x256
#define WK_OFF   262144    // 1024x256
#define WO_OFF   524288    // 256x256 (row o, col c)
#define VT_OFF   589824    // 256x64  (v transposed: [c][n])
#define CVT_TOTAL 606208

// LDS arena (bytes): two TDM weight buffers + aliased staging buffers
#define SM_WBUF0   0               // 64*264*2 = 33792
#define SM_WBUF1   33792           // 33792
#define SM_XFEAT   67584           // xlds during Q/K, featbf afterwards (33792)
#define SM_POOLBF  101376          // 64*72*2 = 9216
#define SM_RQ      110592          // 64*16*4 = 4096
#define SM_POOLED  114688          // 64*64*4 = 16384
#define SM_TOTAL   131072          // 128 KB

#define WMMA(a, b, c, ra, rb)                                                  \
  __builtin_amdgcn_wmma_f32_16x16x32_bf16(false, (a), false, (b), (short)0,    \
                                          (c), (ra), (rb))

__device__ __forceinline__ unsigned short f2bf(float f) {
  unsigned int u = __float_as_uint(f);
  u += 0x7FFFu + ((u >> 16) & 1u);          // round-to-nearest-even
  return (unsigned short)(u >> 16);
}

// One 16x32 (A) / 32x16 (B) bf16 fragment: two b128 loads per lane.
__device__ __forceinline__ v16bf frag16(const unsigned short* p) {
  union { u32x4 u[2]; v16bf v; } f;
  f.u[0] = *(const u32x4*)(p);
  f.u[1] = *(const u32x4*)(p + 16);
  return f.v;
}

// ---------------------------------------------------------------------------
// TDM: async 2D tile load (CH_ROWS x 256 bf16) global -> LDS, with LDS row
// padding of 4 dwords per 128 dwords (row stride 512B -> 528B == XSTR*2).
// Descriptor per CDNA5 ISA (groups 0 and 1, 2D tile).
// ---------------------------------------------------------------------------
__device__ __forceinline__ void tdm_load_chunk(const unsigned short* gsrc,
                                               unsigned int lds_byte_off) {
  unsigned long long ga = (unsigned long long)(size_t)gsrc;
  u32x4 g0;
  g0[0] = 1u;                                    // count=1, user descriptor
  g0[1] = lds_byte_off;                          // lds_addr
  g0[2] = (unsigned int)ga;                      // global_addr[31:0]
  g0[3] = ((unsigned int)(ga >> 32) & 0x01FFFFFFu) | 0x80000000u; // addr[56:32] | type=2
  u32x8 g1;
  g1[0] = (1u << 16)      // data_size = 2 bytes
        | (1u << 20)      // pad_enable
        | (6u << 22)      // pad_interval: every 128 dwords (one 512B row)
        | (3u << 25);     // pad_amount: 4 dwords (16B) -> 528B row stride
  g1[1] = (256u << 16);   // tensor_dim0 = 256 (bits 63:48)
  g1[2] = (1024u << 16);  // tensor_dim0 hi = 0; tensor_dim1 = 1024 (bits 111:80)
  g1[3] = (256u << 16);   // tensor_dim1 hi = 0; tile_dim0 = 256 (bits 127:112)
  g1[4] = (unsigned)CH_ROWS; // tile_dim1 = 64 rows; tile_dim2 = 0
  g1[5] = 256u;           // tensor_dim0_stride[31:0] = 256 elements
  g1[6] = 0u;             // stride hi, tensor_dim1_stride lo
  g1[7] = 0u;
  asm volatile("tensor_load_to_lds %0, %1" : : "s"(g0), "s"(g1) : "memory");
}

// ---------------------------------------------------------------------------
// Pre-pass: fp32 weights -> bf16 workspace (runs once per launch).
// ---------------------------------------------------------------------------
__global__ void cvt_kernel(const float* __restrict__ Wq, const float* __restrict__ Wk,
                           const float* __restrict__ vmat, const float* __restrict__ Wo,
                           unsigned short* __restrict__ ws) {
  int i = blockIdx.x * blockDim.x + threadIdx.x;
  if (i < 262144) { ws[WQ_OFF + i] = f2bf(Wq[i]); return; }
  i -= 262144;
  if (i < 262144) { ws[WK_OFF + i] = f2bf(Wk[i]); return; }
  i -= 262144;
  if (i < 65536)  { ws[WO_OFF + i] = f2bf(Wo[i]); return; }
  i -= 65536;
  if (i < 16384)  {                      // v (64x256) -> vt (256x64)
    int c = i >> 6, n = i & 63;
    ws[VT_OFF + i] = f2bf(vmat[n * C_DIM + c]);
  }
}

// ---------------------------------------------------------------------------
// Main fused kernel: one workgroup (8 wave32) per 64 positions.
// ---------------------------------------------------------------------------
__global__ __launch_bounds__(256)
void fused_kernel(const float* __restrict__ x,
                  const float* __restrict__ bq, const float* __restrict__ bk,
                  const float* __restrict__ bo,
                  const unsigned short* __restrict__ ws,
                  float* __restrict__ out) {
  __shared__ __attribute__((aligned(16))) unsigned char smem[SM_TOTAL];
  unsigned short* wlds     = (unsigned short*)(smem);             // 2 TDM buffers
  unsigned short* xlds     = (unsigned short*)(smem + SM_XFEAT);  // x tile (bf16)
  unsigned short* featbf   = xlds;                                // aliased after K
  unsigned short* pooledbf = (unsigned short*)(smem + SM_POOLBF);
  float*          rq       = (float*)(smem + SM_RQ);
  float*          pooled   = (float*)(smem + SM_POOLED);

  const int tid    = threadIdx.x;
  const int lane   = tid & 31;
  const int w      = tid >> 5;
  const int lanelo = lane & 15;
  const int khalf  = lane >> 4;
  const int ntl    = w & 3;        // n-tile within chunk (4 tiles of 16 rows)
  const int mh     = w >> 2;       // m-tile pair: {0,1} or {2,3}

  const unsigned int smbase = (unsigned int)(size_t)(void*)smem;

  const int p0 = blockIdx.x * POS_TILE;
  const int b  = p0 / L_DIM;
  const int l0 = p0 - b * L_DIM;
  const size_t xbase = (size_t)b * C_DIM * L_DIM + (size_t)l0;

  // ---- init LDS accumulators -------------------------------------------
#pragma unroll
  for (int i = 0; i < 4; ++i)  rq[tid + 256 * i] = 0.f;       // 64x16
#pragma unroll
  for (int i = 0; i < 16; ++i) pooled[tid + 256 * i] = 0.f;   // 64x64

  // ---- stage x tile as bf16, layout [pos][c] ---------------------------
  {
    const int c  = tid >> 2;
    const int pq = (tid & 3) * 16;
    const float* xp = x + xbase + (size_t)c * L_DIM + pq;
#pragma unroll
    for (int j = 0; j < 16; j += 4) {
      float4 xv = *(const float4*)(xp + j);
      xlds[(pq + j + 0) * XSTR + c] = f2bf(xv.x);
      xlds[(pq + j + 1) * XSTR + c] = f2bf(xv.y);
      xlds[(pq + j + 2) * XSTR + c] = f2bf(xv.z);
      xlds[(pq + j + 3) * XSTR + c] = f2bf(xv.w);
    }
  }

  // =======================================================================
  // Q pass then K pass: 16 chunks of 64 weight rows, TDM double-buffered.
  // =======================================================================
  for (int pass = 0; pass < 2; ++pass) {
    const unsigned short* wbf  = ws + (pass ? WK_OFF : WQ_OFF);
    const float*          bias = pass ? bk : bq;

    if (w == 0) tdm_load_chunk(wbf, smbase + SM_WBUF0);          // chunk 0

    for (int ck = 0; ck < N_CHUNK; ++ck) {
      if (w == 0) {
        if (ck < N_CHUNK - 1) {
          tdm_load_chunk(wbf + (size_t)(ck + 1) * CH_ROWS * C_DIM,
                         smbase + ((ck + 1) & 1 ? SM_WBUF1 : SM_WBUF0));
          __builtin_amdgcn_s_wait_tensorcnt(1);   // chunk ck complete
        } else {
          __builtin_amdgcn_s_wait_tensorcnt(0);
        }
      }
      __syncthreads();                            // buf[ck&1] ready for all waves

      const unsigned short* buf  = wlds + (ck & 1 ? (SM_WBUF1 / 2) : 0);
      const unsigned short* brow = buf + (ntl * 16 + lanelo) * XSTR + khalf * 8;
      v8f a0 = {}, a1 = {};
#pragma unroll
      for (int kt = 0; kt < 8; ++kt) {
        v16bf fb  = frag16(brow + kt * 32);
        v16bf fa0 = frag16(&xlds[((mh * 2 + 0) * 16 + lanelo) * XSTR + kt * 32 + khalf * 8]);
        v16bf fa1 = frag16(&xlds[((mh * 2 + 1) * 16 + lanelo) * XSTR + kt * 32 + khalf * 8]);
        a0 = WMMA(fa0, fb, a0, false, false);
        a1 = WMMA(fa1, fb, a1, false, true);      // reuse_b: same B as previous
      }

      const int obase = ck * CH_ROWS + ntl * 16;
      const float bv  = bias[obase + lanelo];
      if (pass == 0) {
        // Q: relu then reduce over n_mem into rq[pos][s]
#pragma unroll
        for (int t2 = 0; t2 < 2; ++t2) {
          v8f a = t2 ? a1 : a0;
#pragma unroll
          for (int vv = 0; vv < 8; ++vv) {
            float q = fmaxf(a[vv] + bv, 0.f);
            const int pos = (mh * 2 + t2) * 16 + vv + khalf * 8;
            atomicAdd(&rq[pos * SQF + lanelo], q);       // ds_add_f32
          }
        }
      } else {
        // K: fold relu(K) against rq, sum over s (16 lanes) -> pooled
        const int nidx = obase >> 4;                     // n_mem index
#pragma unroll
        for (int t2 = 0; t2 < 2; ++t2) {
          v8f a = t2 ? a1 : a0;
#pragma unroll
          for (int vv = 0; vv < 8; ++vv) {
            const int pos = (mh * 2 + t2) * 16 + vv + khalf * 8;
            float kq = fmaxf(a[vv] + bv, 0.f);
            // scale = 1/sqrt(16); mean over m = /64 => 1/256 total
            float t = kq * rq[pos * SQF + lanelo] * (1.0f / 256.0f);
            t += __shfl_xor(t, 1, 32);
            t += __shfl_xor(t, 2, 32);
            t += __shfl_xor(t, 4, 32);
            t += __shfl_xor(t, 8, 32);
            if (lanelo == 0) pooled[pos * NMEM + nidx] = t;
          }
        }
      }
      __syncthreads();     // all waves done with buf[ck&1] before TDM overwrites
    }
  }

  // ---- pooled -> bf16 (A matrix of feat GEMM) --------------------------
  __builtin_prefetch(ws + WO_OFF + (size_t)((w * 2) * 16 + lanelo) * C_DIM, 0, 0);
#pragma unroll
  for (int i = 0; i < 16; ++i) {
    const int idx = tid + 256 * i;                 // 4096 elements
    const int pos = idx >> 6, n = idx & 63;
    pooledbf[pos * PSTR + n] = f2bf(pooled[idx]);
  }
  __syncthreads();

  // ---- feat GEMM: feat(64x256) = pooled(64x64) @ v(64x256) -------------
  {
    const unsigned short* vtbf = ws + VT_OFF;
    v8f fc[2][4];
#pragma unroll
    for (int nl = 0; nl < 2; ++nl)
#pragma unroll
      for (int mt = 0; mt < 4; ++mt) fc[nl][mt] = {};
    const unsigned short* vb0 = vtbf + ((w * 2 + 0) * 16 + lanelo) * NMEM + khalf * 8;
    const unsigned short* vb1 = vtbf + ((w * 2 + 1) * 16 + lanelo) * NMEM + khalf * 8;
#pragma unroll
    for (int kt = 0; kt < 2; ++kt) {
      v16bf fb0 = frag16(vb0 + kt * 32);
      v16bf fb1 = frag16(vb1 + kt * 32);
#pragma unroll
      for (int mt = 0; mt < 4; ++mt) {
        v16bf fa = frag16(&pooledbf[(mt * 16 + lanelo) * PSTR + kt * 32 + khalf * 8]);
        fc[0][mt] = WMMA(fa, fb0, fc[0][mt], false, false);
        fc[1][mt] = WMMA(fa, fb1, fc[1][mt], true, false);  // reuse_a: same A
      }
    }
    __syncthreads();   // everyone done reading pooledbf/xlds before featbf aliases
    // feat -> bf16 LDS, layout [pos][c]  (featbf aliases xlds; xlds is dead)
#pragma unroll
    for (int nl = 0; nl < 2; ++nl) {
      const int cb = (w * 2 + nl) * 16 + lanelo;
#pragma unroll
      for (int mt = 0; mt < 4; ++mt)
#pragma unroll
        for (int vv = 0; vv < 8; ++vv) {
          const int pos = mt * 16 + vv + khalf * 8;
          featbf[pos * XSTR + cb] = f2bf(fc[nl][mt][vv]);
        }
    }
  }
  __syncthreads();

  // ---- out GEMM: out(64x256) = feat(64x256) @ Wo^T + bo + x ------------
  {
    const unsigned short* wobf = ws + WO_OFF;
    v8f oc[2][4];
#pragma unroll
    for (int nl = 0; nl < 2; ++nl)
#pragma unroll
      for (int mt = 0; mt < 4; ++mt) oc[nl][mt] = {};
    const int oo0 = (w * 2 + 0) * 16 + lanelo;
    const int oo1 = (w * 2 + 1) * 16 + lanelo;
    const unsigned short* wb0 = wobf + (size_t)oo0 * C_DIM + khalf * 8;
    const unsigned short* wb1 = wobf + (size_t)oo1 * C_DIM + khalf * 8;
#pragma unroll
    for (int kt = 0; kt < 8; ++kt) {
      v16bf fb0 = frag16(wb0 + kt * 32);
      v16bf fb1 = frag16(wb1 + kt * 32);
#pragma unroll
      for (int mt = 0; mt < 4; ++mt) {
        v16bf fa = frag16(&featbf[(mt * 16 + lanelo) * XSTR + kt * 32 + khalf * 8]);
        oc[0][mt] = WMMA(fa, fb0, oc[0][mt], false, false);
        oc[1][mt] = WMMA(fa, fb1, oc[1][mt], true, false);  // reuse_a: same A
      }
    }
    // epilogue: + bo + residual x, store fp32
    const float bo0 = bo[oo0];
    const float bo1 = bo[oo1];
    const size_t g0 = (size_t)b * C_DIM * L_DIM + (size_t)oo0 * L_DIM + l0;
    const size_t g1 = (size_t)b * C_DIM * L_DIM + (size_t)oo1 * L_DIM + l0;
#pragma unroll
    for (int mt = 0; mt < 4; ++mt)
#pragma unroll
      for (int vv = 0; vv < 8; ++vv) {
        const int pos = mt * 16 + vv + khalf * 8;
        out[g0 + pos] = oc[0][mt][vv] + bo0 + x[g0 + pos];
        out[g1 + pos] = oc[1][mt][vv] + bo1 + x[g1 + pos];
      }
  }
}

// ---------------------------------------------------------------------------
extern "C" void kernel_launch(void* const* d_in, const int* in_sizes, int n_in,
                              void* d_out, int out_size, void* d_ws, size_t ws_size,
                              hipStream_t stream) {
  const float* x    = (const float*)d_in[0];
  const float* Wq   = (const float*)d_in[1];
  const float* bq   = (const float*)d_in[2];
  const float* Wk   = (const float*)d_in[3];
  const float* bk   = (const float*)d_in[4];
  const float* vmat = (const float*)d_in[5];
  const float* Wo   = (const float*)d_in[6];
  const float* bo   = (const float*)d_in[7];
  float* outp = (float*)d_out;
  unsigned short* wsp = (unsigned short*)d_ws;

  // 1) weights fp32 -> bf16 (and v transposed) into workspace
  cvt_kernel<<<CVT_TOTAL / 256, 256, 0, stream>>>(Wq, Wk, vmat, Wo, wsp);

  // 2) fused block: 16384 positions / 64 per WG = 256 workgroups
  fused_kernel<<<(4 * L_DIM) / POS_TILE, 256, 0, stream>>>(x, bq, bk, bo, wsp, outp);
}
